// ComAttention_79491254714605
// MI455X (gfx1250) — matrix operand order repaired and checked by reading, not compiled, not run on hardware
//
#include <hip/hip_runtime.h>
#include <hip/hip_bf16.h>
#include <math.h>

typedef __attribute__((ext_vector_type(16))) _Float16 v16h;
typedef __attribute__((ext_vector_type(8)))  _Float16 v8h;
typedef __attribute__((ext_vector_type(8)))  float    v8f;

#define B_   4
#define S_   2048
#define D_   32
#define NQT  (S_/16)
#define RS8  0.35355339059327373f   /* 1/sqrt(8) */
#define NEG9 -1.0e9f

/* ---- workspace layout (bytes) ---- */
#define HDR_OFF   0                          /* float w_eff[4], b_eff       */
#define PROJ_OFF  256
#define PROJ_SZ   (B_*S_*D_*2)               /* 512 KB per f16 projection   */
#define QG_OFF   (PROJ_OFF + 0*PROJ_SZ)
#define KG_OFF   (PROJ_OFF + 1*PROJ_SZ)
#define QP_OFF   (PROJ_OFF + 2*PROJ_SZ)
#define KP_OFF   (PROJ_OFF + 3*PROJ_SZ)
#define QN_OFF   (PROJ_OFF + 4*PROJ_SZ)
#define KN_OFF   (PROJ_OFF + 5*PROJ_SZ)
#define VPT_OFF  (PROJ_OFF + 6*PROJ_SZ)
#define VNT_OFF  (PROJ_OFF + 7*PROJ_SZ)
#define ATT_OFF  (PROJ_OFF + 8*PROJ_SZ)      /* f32 attnP then attnN, 1MB each */
#define ATT_N    (B_*S_*D_)
#define POUT_OFF (ATT_OFF + 2*ATT_N*4)       /* f32 poutP then poutN        */

__device__ inline v16h mk16(v8h lo, v8h hi) {
  v16h r;
  #pragma unroll
  for (int i = 0; i < 8; ++i) { r[i] = lo[i]; r[8 + i] = hi[i]; }
  return r;
}
__device__ inline v16h ldB(const _Float16* p) {
  return mk16(*(const v8h*)p, *(const v8h*)(p + 16));
}
__device__ inline v8f wmma16(v16h a, v16h b, v8f c) {
  return __builtin_amdgcn_wmma_f32_16x16x32_f16(false, a, false, b, (short)0, c,
                                                false, false);
}

/* ---- stage 0: collapse conv stack to w_eff[4], b_eff ---- */
__global__ void gconst_kernel(const float* f1w, const float* f1b,
                              const float* f2w, const float* f2b,
                              const float* f3w, const float* f3b, float* hdr) {
  if (blockIdx.x == 0 && threadIdx.x == 0) {
    for (int h = 0; h < 4; ++h) {
      float s = 0.f;
      for (int j = 0; j < 4; ++j)
        for (int i = 0; i < 8; ++i) s += f3w[j] * f2w[j * 8 + i] * f1w[i * 4 + h];
      hdr[h] = s;
    }
    float be = f3b[0];
    for (int j = 0; j < 4; ++j) {
      float t = f2b[j];
      for (int i = 0; i < 8; ++i) t += f2w[j * 8 + i] * f1b[i];
      be += f3w[j] * t;
    }
    hdr[4] = be;
  }
}

/* ---- stage 1: 8 fused projections to f16 (scales/biases folded) ---- */
__global__ void proj_kernel(const float* __restrict__ feat,
                            const float* qw, const float* qb,
                            const float* kw, const float* kb,
                            const float* paw, const float* pab,
                            const float* naw, const float* nab,
                            const float* hdr,
                            _Float16* Qg, _Float16* Kg, _Float16* Qp, _Float16* Kp,
                            _Float16* Qn, _Float16* Kn, _Float16* VpT, _Float16* VnT) {
  int tid = blockIdx.x * blockDim.x + threadIdx.x;
  if (tid >= 8 * B_ * S_) return;
  int proj = tid >> 13;         /* B_*S_ = 8192 */
  int pos  = tid & 8191;
  const float* f = feat + pos * D_;
  const float* w = qw; const float* bias = qb;
  float scale = 1.f; _Float16* dst = Qg; bool transposed = false; bool gatew = false;
  switch (proj) {
    case 0: w = qw;        bias = qb;       dst = Qg;  scale = RS8; gatew = true; break;
    case 1: w = kw;        bias = kb;       dst = Kg;  break;
    case 2: w = paw;       bias = pab;      dst = Qp;  scale = RS8; break;
    case 3: w = paw+1024;  bias = pab+32;   dst = Kp;  break;
    case 4: w = paw+2048;  bias = pab+64;   dst = VpT; transposed = true; break;
    case 5: w = naw;       bias = nab;      dst = Qn;  scale = RS8; break;
    case 6: w = naw+1024;  bias = nab+32;   dst = Kn;  break;
    case 7: w = naw+2048;  bias = nab+64;   dst = VnT; transposed = true; break;
  }
  int bidx = pos >> 11, srow = pos & 2047;
  for (int o = 0; o < D_; ++o) {
    float a = bias[o];
    #pragma unroll
    for (int i = 0; i < D_; ++i) a += f[i] * w[o * D_ + i];
    a *= scale;
    if (gatew) a *= hdr[o >> 3];
    if (transposed) dst[(bidx * D_ + o) * S_ + srow] = (_Float16)a;
    else            dst[pos * D_ + o] = (_Float16)a;
  }
}

/* ---- stage 2: fused dual flash attention (gate-sign routed) ---- */
__global__ __launch_bounds__(32)
void attn_kernel(const _Float16* __restrict__ Qg, const _Float16* __restrict__ Kg,
                 const _Float16* __restrict__ Qp, const _Float16* __restrict__ Kp,
                 const _Float16* __restrict__ Qn, const _Float16* __restrict__ Kn,
                 const _Float16* __restrict__ VpT, const _Float16* __restrict__ VnT,
                 const float* __restrict__ dmask, const float* __restrict__ hdr,
                 float* __restrict__ attP, float* __restrict__ attN) {
  __shared__ _Float16 Pst[4][16][32];          /* P tiles staged as f16, 4 KB */
  const int wid  = blockIdx.x;
  const int pn   = wid >> 9;                   /* 512 = B_*NQT waves per branch */
  const int b    = (wid >> 7) & 3;
  const int qt   = wid & 127;
  const int lane = threadIdx.x;
  const int g = lane >> 4, m = lane & 15;
  const float beff = hdr[4];

  const _Float16* Qx = pn ? Qn  : Qp;
  const _Float16* Kx = pn ? Kn  : Kp;
  const _Float16* VT = pn ? VnT : VpT;
  float*        outp = pn ? attN : attP;

  /* A fragments: lane group g holds dims {8g..8g+7, 16+8g..16+8g+7} of row m */
  const int qrow = qt * 16 + m;
  const _Float16* qptr = Qx + (size_t)(b * S_ + qrow) * D_;
  v8h qlo = *(const v8h*)(qptr + g * 8);
  v8h qhi = *(const v8h*)(qptr + 16 + g * 8);
  v16h Ah[4];
  #pragma unroll
  for (int h = 0; h < 4; ++h) {
    v16h a = (v16h)(_Float16)0;                /* head h = dims [8h,8h+8)     */
    if (h == g) {
      #pragma unroll
      for (int i = 0; i < 8; ++i) a[i] = qlo[i];
    }
    if (h == g + 2) {
      #pragma unroll
      for (int i = 0; i < 8; ++i) a[8 + i] = qhi[i];
    }
    Ah[h] = a;
  }
  const _Float16* gq = Qg + (size_t)(b * S_ + qrow) * D_;
  v16h Ag = mk16(*(const v8h*)(gq + g * 8), *(const v8h*)(gq + 16 + g * 8));

  v8f acc[4], mrow[4], lrow[4];
  #pragma unroll
  for (int h = 0; h < 4; ++h) {
    #pragma unroll
    for (int r = 0; r < 8; ++r) { acc[h][r] = 0.f; mrow[h][r] = -INFINITY; lrow[h][r] = 0.f; }
  }

  const v8f zacc = (v8f)0.0f;

  for (int kc = 0; kc < S_; kc += 32) {
    /* B fragments (all contiguous 16B loads) */
    v16h Bg0 = ldB(Kg + (size_t)(b * S_ + kc +      m) * D_ + g * 8);
    v16h Bg1 = ldB(Kg + (size_t)(b * S_ + kc + 16 + m) * D_ + g * 8);
    v16h Bk0 = ldB(Kx + (size_t)(b * S_ + kc +      m) * D_ + g * 8);
    v16h Bk1 = ldB(Kx + (size_t)(b * S_ + kc + 16 + m) * D_ + g * 8);
    v16h Bv0 = ldB(VT + (size_t)(b * D_ +      m) * S_ + kc + g * 8);
    v16h Bv1 = ldB(VT + (size_t)(b * D_ + 16 + m) * S_ + kc + g * 8);

    /* gate tiles: sign routes (q,k) to p or n */
    v8f G0 = wmma16(Ag, Bg0, zacc);
    v8f G1 = wmma16(Ag, Bg1, zacc);

    bool ok0[8], ok1[8];
    #pragma unroll
    for (int r = 0; r < 8; ++r) {
      int q = qt * 16 + r + 8 * g;
      float d0 = dmask[(size_t)(b * S_ + q) * S_ + kc + m];
      float d1 = dmask[(size_t)(b * S_ + q) * S_ + kc + 16 + m];
      float g0 = G0[r] + beff, g1 = G1[r] + beff;
      bool s0 = pn ? (g0 <= 0.f) : (g0 > 0.f);
      bool s1 = pn ? (g1 <= 0.f) : (g1 > 0.f);
      ok0[r] = s0 && (d0 != 0.f);
      ok1[r] = s1 && (d1 != 0.f);
    }

    #pragma unroll
    for (int h = 0; h < 4; ++h) {
      v8f S0 = wmma16(Ah[h], Bk0, zacc);
      v8f S1 = wmma16(Ah[h], Bk1, zacc);
      v8f t;
      #pragma unroll
      for (int r = 0; r < 8; ++r) {
        float a = ok0[r] ? S0[r] : NEG9;
        float c = ok1[r] ? S1[r] : NEG9;
        S0[r] = a; S1[r] = c; t[r] = fmaxf(a, c);
      }
      /* row max across the 16-lane half (N dim of the C tile) */
      #pragma unroll
      for (int mk = 1; mk < 16; mk <<= 1) {
        #pragma unroll
        for (int r = 0; r < 8; ++r) t[r] = fmaxf(t[r], __shfl_xor(t[r], mk));
      }
      #pragma unroll
      for (int r = 0; r < 8; ++r) {
        float mn = fmaxf(mrow[h][r], t[r]);
        float sc = __expf(mrow[h][r] - mn);
        float p0 = __expf(S0[r] - mn);
        float p1 = __expf(S1[r] - mn);
        lrow[h][r] = lrow[h][r] * sc + p0 + p1;
        acc[h][r] *= sc;
        mrow[h][r] = mn;
        int q = r + 8 * g;
        Pst[h][q][m]      = (_Float16)p0;
        Pst[h][q][16 + m] = (_Float16)p1;
      }
    }
    __syncthreads();
    /* P·V: B half 0 = dims 0-15 (heads 0,1), half 1 = dims 16-31 (heads 2,3) */
    #pragma unroll
    for (int h = 0; h < 4; ++h) {
      v16h Ap = mk16(*(const v8h*)&Pst[h][m][g * 8],
                     *(const v8h*)&Pst[h][m][16 + g * 8]);
      acc[h] = wmma16(Ap, (h < 2) ? Bv0 : Bv1, acc[h]);
    }
    __syncthreads();
  }

  /* epilogue: reduce l across the half, normalize, store the 8 valid cols */
  #pragma unroll
  for (int h = 0; h < 4; ++h) {
    v8f L = lrow[h];
    #pragma unroll
    for (int mk = 1; mk < 16; mk <<= 1) {
      #pragma unroll
      for (int r = 0; r < 8; ++r) L[r] += __shfl_xor(L[r], mk);
    }
    if ((m >> 3) == (h & 1)) {
      #pragma unroll
      for (int r = 0; r < 8; ++r) {
        int q = qt * 16 + r + 8 * g;
        int dim = (h >> 1) * 16 + m;
        outp[(size_t)(b * S_ + q) * D_ + dim] = acc[h][r] / L[r];
      }
    }
  }
}

/* ---- stage 3a: MHA output projections ---- */
__global__ void outproj_kernel(const float* __restrict__ attP, const float* __restrict__ attN,
                               const float* paw, const float* pab,
                               const float* naw, const float* nab,
                               float* poutP, float* poutN) {
  int tid = blockIdx.x * blockDim.x + threadIdx.x;
  if (tid >= 2 * ATT_N) return;
  int pn = tid >> 18;                     /* ATT_N = 262144 = 2^18 */
  int r  = tid & (ATT_N - 1);
  int pos = r >> 5, o = r & 31;
  const float* att = (pn ? attN : attP) + pos * D_;
  const float* w = (pn ? naw : paw) + 3 * 1024 + o * D_;
  float a = (pn ? nab : pab)[96 + o];
  #pragma unroll
  for (int i = 0; i < D_; ++i) a += att[i] * w[i];
  (pn ? poutN : poutP)[pos * D_ + o] = a;
}

/* ---- stage 3b: value/gate projections + 2-way softmax combine ---- */
__global__ void gate_kernel(const float* __restrict__ poutP, const float* __restrict__ poutN,
                            const float* vpw, const float* vpb,
                            const float* vnw, const float* vnb,
                            const float* gpw, const float* gpb,
                            float* __restrict__ out) {
  int tid = blockIdx.x * blockDim.x + threadIdx.x;
  if (tid >= ATT_N) return;
  int pos = tid >> 5, o = tid & 31;
  const float* pr = poutP + pos * D_;
  const float* nr = poutN + pos * D_;
  float vp = vpb[o], vn = vnb[o], ep = gpb[o], en = gpb[o];
  #pragma unroll
  for (int i = 0; i < D_; ++i) {
    vp += pr[i] * vpw[o * D_ + i];
    vn += nr[i] * vnw[o * D_ + i];
    ep += pr[i] * gpw[o * D_ + i];   /* reference uses gp_w for BOTH gates */
    en += nr[i] * gpw[o * D_ + i];
  }
  float mx = fmaxf(ep, en);
  float a = __expf(ep - mx), c = __expf(en - mx);
  out[tid] = (vp * a + vn * c) / (a + c);
}

extern "C" void kernel_launch(void* const* d_in, const int* in_sizes, int n_in,
                              void* d_out, int out_size, void* d_ws, size_t ws_size,
                              hipStream_t stream) {
  const float* feat  = (const float*)d_in[0];
  const float* dmask = (const float*)d_in[1];
  const float* qw  = (const float*)d_in[2];
  const float* qb  = (const float*)d_in[3];
  const float* kw  = (const float*)d_in[4];
  const float* kb  = (const float*)d_in[5];
  const float* f1w = (const float*)d_in[6];
  const float* f1b = (const float*)d_in[7];
  const float* f2w = (const float*)d_in[8];
  const float* f2b = (const float*)d_in[9];
  const float* f3w = (const float*)d_in[10];
  const float* f3b = (const float*)d_in[11];
  const float* paw = (const float*)d_in[12];
  const float* pab = (const float*)d_in[13];
  const float* naw = (const float*)d_in[14];
  const float* nab = (const float*)d_in[15];
  const float* vpw = (const float*)d_in[16];
  const float* vpb = (const float*)d_in[17];
  const float* vnw = (const float*)d_in[18];
  const float* vnb = (const float*)d_in[19];
  const float* gpw = (const float*)d_in[20];
  const float* gpb = (const float*)d_in[21];

  char* ws = (char*)d_ws;
  float*    hdr = (float*)(ws + HDR_OFF);
  _Float16* Qg  = (_Float16*)(ws + QG_OFF);
  _Float16* Kg  = (_Float16*)(ws + KG_OFF);
  _Float16* Qp  = (_Float16*)(ws + QP_OFF);
  _Float16* Kp  = (_Float16*)(ws + KP_OFF);
  _Float16* Qn  = (_Float16*)(ws + QN_OFF);
  _Float16* Kn  = (_Float16*)(ws + KN_OFF);
  _Float16* VpT = (_Float16*)(ws + VPT_OFF);
  _Float16* VnT = (_Float16*)(ws + VNT_OFF);
  float* attP  = (float*)(ws + ATT_OFF);
  float* attN  = attP + ATT_N;
  float* poutP = (float*)(ws + POUT_OFF);
  float* poutN = poutP + ATT_N;

  gconst_kernel<<<1, 32, 0, stream>>>(f1w, f1b, f2w, f2b, f3w, f3b, hdr);
  proj_kernel<<<(8 * B_ * S_ + 255) / 256, 256, 0, stream>>>(
      feat, qw, qb, kw, kb, paw, pab, naw, nab, hdr,
      Qg, Kg, Qp, Kp, Qn, Kn, VpT, VnT);
  attn_kernel<<<2 * B_ * NQT, 32, 0, stream>>>(
      Qg, Kg, Qp, Kp, Qn, Kn, VpT, VnT, dmask, hdr, attP, attN);
  outproj_kernel<<<(2 * ATT_N + 255) / 256, 256, 0, stream>>>(
      attP, attN, paw, pab, naw, nab, poutP, poutN);
  gate_kernel<<<(ATT_N + 255) / 256, 256, 0, stream>>>(
      poutP, poutN, vpw, vpb, vnw, vnb, gpw, gpb, (float*)d_out);
}
